// FunctorF_29351806501385
// MI455X (gfx1250) — matrix operand order repaired and checked by reading, compiled, not run on hardware
//
#include <hip/hip_runtime.h>
#include <hip/hip_bf16.h>

// ---------------------------------------------------------------------------
// PointNet++-style SA network for gfx1250 (MI455X).
// Linear layers via v_wmma_f32_16x16x32_f16 (f16 in, f32 accum), weights
// pre-transposed so both GEMM operands stage into LDS with 128-bit ops.
// Double-buffered LDS pipeline: one barrier per K-step.
// BN stats via atomic partial sums; segment_max via integer atomicMax on
// non-negative float bit patterns (valid since msg >= 0 after ReLU).
// ---------------------------------------------------------------------------

#define NPTS 8192
#define KNN  16
#define NEDGE (NPTS * KNN)
#define EPSBN 1e-5f

typedef _Float16 v16h __attribute__((ext_vector_type(16)));
typedef float    v8f  __attribute__((ext_vector_type(8)));

// ---------------------------------------------------------------------------
// kNN: brute force, thread per point, top-(K+1) insertion sort, drop self.
// ---------------------------------------------------------------------------
__global__ void knn_kernel(const float* __restrict__ pos, int* __restrict__ idx)
{
    int i = blockIdx.x * blockDim.x + threadIdx.x;
    if (i >= NPTS) return;
    const float px = pos[i * 3 + 0], py = pos[i * 3 + 1], pz = pos[i * 3 + 2];
    float bd[KNN + 1];
    int   bi[KNN + 1];
#pragma unroll
    for (int t = 0; t <= KNN; ++t) { bd[t] = 3.4e38f; bi[t] = 0; }
    for (int j = 0; j < NPTS; ++j) {
        float dx = px - pos[j * 3 + 0];
        float dy = py - pos[j * 3 + 1];
        float dz = pz - pos[j * 3 + 2];
        float d2 = dx * dx + dy * dy + dz * dz;
        if (d2 < bd[KNN]) {
            int t = KNN;
            while (t > 0 && bd[t - 1] > d2) { bd[t] = bd[t - 1]; bi[t] = bi[t - 1]; --t; }
            bd[t] = d2; bi[t] = j;
        }
    }
    for (int k = 0; k < KNN; ++k) idx[i * KNN + k] = bi[k + 1];  // drop self
}

// ---------------------------------------------------------------------------
// Embed linear: Y[N][64] = pos @ W(3x64) + b
// ---------------------------------------------------------------------------
__global__ void embed_kernel(const float* __restrict__ pos, const float* __restrict__ W,
                             const float* __restrict__ b, float* __restrict__ Y, int total)
{
    int gid = blockIdx.x * blockDim.x + threadIdx.x;
    if (gid >= total) return;
    int c = gid & 63, i = gid >> 6;
    float v = b[c];
    v += pos[i * 3 + 0] * W[0 * 64 + c];
    v += pos[i * 3 + 1] * W[1 * 64 + c];
    v += pos[i * 3 + 2] * W[2 * 64 + c];
    Y[gid] = v;
}

// ---------------------------------------------------------------------------
// Gather + concat + pad -> f16 edge features: Ah[E][ICp]
// ---------------------------------------------------------------------------
__global__ void pack_sa_kernel(const float* __restrict__ x, const float* __restrict__ pos,
                               const int* __restrict__ idx, _Float16* __restrict__ Ah,
                               int inch, int ICp, long long total)
{
    long long gid = (long long)blockIdx.x * blockDim.x + threadIdx.x;
    if (gid >= total) return;
    int c = (int)(gid % ICp);
    long long e = gid / ICp;
    int row = (int)(e / KNN);
    float v = 0.0f;
    if (c < inch) {
        v = x[(size_t)row * inch + c];
    } else if (c < inch + 3) {
        int col = idx[e];
        int d = c - inch;
        v = pos[row * 3 + d] - pos[col * 3 + d];
    }
    Ah[gid] = (_Float16)v;
}

// ---------------------------------------------------------------------------
// Weight convert + TRANSPOSE + K-pad: W[IC][OC] (f32) -> Wt[OC][ICp] (f16).
// ---------------------------------------------------------------------------
__global__ void cvt_wT_f16_kernel(const float* __restrict__ W, _Float16* __restrict__ Wt,
                                  int IC, int ICp, int OC, long long total)
{
    long long gid = (long long)blockIdx.x * blockDim.x + threadIdx.x;
    if (gid >= total) return;
    int k = (int)(gid % ICp);
    int c = (int)(gid / ICp);
    Wt[gid] = (_Float16)(k < IC ? W[(size_t)k * OC + c] : 0.0f);
}

// ---------------------------------------------------------------------------
// Plain f32 -> f16 row-major convert (activations for head GEMM).
// ---------------------------------------------------------------------------
__global__ void cvt_f16_kernel(const float* __restrict__ src, _Float16* __restrict__ dst,
                               long long total)
{
    long long gid = (long long)blockIdx.x * blockDim.x + threadIdx.x;
    if (gid >= total) return;
    dst[gid] = (_Float16)src[gid];
}

// ---------------------------------------------------------------------------
// WMMA GEMM: C[M][Nc] = A[M][Kp](f16) @ Bt[Nc][Kp](f16, transposed) + bias.
// Block tile 128x64, 8 waves; wave w owns a 32x32 output patch:
//   m-group = w&3 (rows mg*32..+31), n-group = w>>2 (cols ng*32..+31)
//   -> 4 accumulators, 4 v_wmma per 32-K step, 2 A-frags + 2 B-frags.
// Double-buffered LDS (rows padded to 40 halfs for bank spread), single
// barrier per K-step; next tile prefetched to registers during compute.
// ---------------------------------------------------------------------------
__global__ __launch_bounds__(256) void gemm_wmma_kernel(
    const _Float16* __restrict__ A, const _Float16* __restrict__ Bt,
    const float* __restrict__ bias, float* __restrict__ C,
    int M, int Kp, int Nc)
{
    __shared__ _Float16 As[2][128][40];
    __shared__ _Float16 Bs[2][64][40];

    const int m0 = blockIdx.x * 128;
    const int n0 = blockIdx.y * 64;
    const int tid  = threadIdx.x;
    const int lane = tid & 31;
    const int wave = tid >> 5;
    const int mg = wave & 3;
    const int ng = wave >> 2;
    const int lane16 = lane & 15;
    const bool hi = lane >= 16;

    // Staging assignments (per thread): A: 16 halfs of one row; B: 8 halfs.
    const int arow  = tid >> 1;           // 0..127
    const int akoff = (tid & 1) * 16;     // 0 or 16
    const int brow  = tid >> 2;           // 0..63
    const int bkoff = (tid & 3) * 8;      // 0,8,16,24

    const _Float16* aptr = A  + (size_t)(m0 + arow) * Kp + akoff;
    const _Float16* bptr = Bt + (size_t)(n0 + brow) * Kp + bkoff;

    v8f acc00 = {}, acc01 = {}, acc10 = {}, acc11 = {};

    // Preload tile 0 into registers.
    uint4 ra0 = *(const uint4*)(aptr);
    uint4 ra1 = *(const uint4*)(aptr + 8);
    uint4 rb  = *(const uint4*)(bptr);

    int buf = 0;
    for (int k0 = 0; k0 < Kp; k0 += 32) {
        // Commit staged registers to LDS buffer `buf`.
        *(uint4*)&As[buf][arow][akoff]     = ra0;
        *(uint4*)&As[buf][arow][akoff + 8] = ra1;
        *(uint4*)&Bs[buf][brow][bkoff]     = rb;
        __syncthreads();

        // Prefetch next tile into registers (overlaps with WMMA below).
        if (k0 + 32 < Kp) {
            ra0 = *(const uint4*)(aptr + k0 + 32);
            ra1 = *(const uint4*)(aptr + k0 + 40);
            rb  = *(const uint4*)(bptr + k0 + 32);
            if (k0 + 64 < Kp) {
                __builtin_prefetch(aptr + k0 + 64, 0, 0);
                __builtin_prefetch(bptr + k0 + 64, 0, 0);
            }
        }

        // Fragments per CDNA5 ISA 7.12.2.
        // A 16x32: lanes 0-15 row M=lane, halfs K 0..7,16..23; lanes 16-31
        // same rows, halfs K 8..15,24..31.
        const int selA = hi ? 8 : 0;
        const int kB   = hi ? 8 : 0;

        union { v16h v; uint4 u[2]; } ua0, ua1, ub0, ub1;
        {
            const int rowA0 = mg * 32 + lane16;
            const int rowA1 = rowA0 + 16;
            ua0.u[0] = *(const uint4*)&As[buf][rowA0][selA];
            ua0.u[1] = *(const uint4*)&As[buf][rowA0][selA + 16];
            ua1.u[0] = *(const uint4*)&As[buf][rowA1][selA];
            ua1.u[1] = *(const uint4*)&As[buf][rowA1][selA + 16];
        }
        // B 32x16: lane holds column N=lane%16; lanes<16 K 0..7,16..23, lanes>=16
        // K 8..15,24..31 (same order as A).
        {
            const int colB0 = ng * 32 + lane16;
            const int colB1 = colB0 + 16;
            ub0.u[0] = *(const uint4*)&Bs[buf][colB0][kB];
            ub0.u[1] = *(const uint4*)&Bs[buf][colB0][kB + 16];
            ub1.u[0] = *(const uint4*)&Bs[buf][colB1][kB];
            ub1.u[1] = *(const uint4*)&Bs[buf][colB1][kB + 16];
        }

        acc00 = __builtin_amdgcn_wmma_f32_16x16x32_f16(false, ua0.v, false, ub0.v,
                                                       (short)0, acc00, false, false);
        acc01 = __builtin_amdgcn_wmma_f32_16x16x32_f16(false, ua0.v, false, ub1.v,
                                                       (short)0, acc01, false, false);
        acc10 = __builtin_amdgcn_wmma_f32_16x16x32_f16(false, ua1.v, false, ub0.v,
                                                       (short)0, acc10, false, false);
        acc11 = __builtin_amdgcn_wmma_f32_16x16x32_f16(false, ua1.v, false, ub1.v,
                                                       (short)0, acc11, false, false);
        buf ^= 1;
    }

    // C layout: VGPR r -> M = r (lanes<16) / 8+r (lanes>=16); N = lane%16.
    const int rb0 = m0 + mg * 32 + (hi ? 8 : 0);
    const int c0  = n0 + ng * 32 + lane16;
    const int c1  = c0 + 16;
    const float bb0 = bias[c0];
    const float bb1 = bias[c1];
#pragma unroll
    for (int r = 0; r < 8; ++r) {
        C[(size_t)(rb0 + r) * Nc + c0]      = acc00[r] + bb0;
        C[(size_t)(rb0 + r) * Nc + c1]      = acc01[r] + bb1;
        C[(size_t)(rb0 + 16 + r) * Nc + c0] = acc10[r] + bb0;
        C[(size_t)(rb0 + 16 + r) * Nc + c1] = acc11[r] + bb1;
    }
}

// ---------------------------------------------------------------------------
// BN stats: per-channel sum & sum-of-squares (atomic partial sums).
// ---------------------------------------------------------------------------
__global__ void bn_stats_kernel(const float* __restrict__ Y, float* __restrict__ sums,
                                int M, int C)
{
    int gid = blockIdx.x * blockDim.x + threadIdx.x;
    int c  = gid % C;
    int rb = gid / C;
    int r0 = rb * 256;
    if (r0 >= M) return;
    int r1 = r0 + 256; if (r1 > M) r1 = M;
    float s = 0.0f, s2 = 0.0f;
    for (int r = r0; r < r1; ++r) {
        float v = Y[(size_t)r * C + c];
        s += v; s2 += v * v;
    }
    atomicAdd(&sums[c], s);
    atomicAdd(&sums[C + c], s2);
}

__global__ void zero_kernel(float* __restrict__ p, long long n)
{
    long long gid = (long long)blockIdx.x * blockDim.x + threadIdx.x;
    if (gid < n) p[gid] = 0.0f;
}

// BN + ReLU, f32 output
__global__ void bn_relu_f32_kernel(const float* __restrict__ Y, const float* __restrict__ sums,
                                   const float* __restrict__ g, const float* __restrict__ beta,
                                   float* __restrict__ out, long long total, int C, float invM)
{
    long long gid = (long long)blockIdx.x * blockDim.x + threadIdx.x;
    if (gid >= total) return;
    int c = (int)(gid % C);
    float m  = sums[c] * invM;
    float vv = sums[C + c] * invM - m * m;
    float s  = g[c] * rsqrtf(vv + EPSBN);
    float v  = (Y[gid] - m) * s + beta[c];
    out[gid] = fmaxf(v, 0.0f);
}

// BN + ReLU, f16 output (feeds next WMMA GEMM)
__global__ void bn_relu_f16_kernel(const float* __restrict__ Y, const float* __restrict__ sums,
                                   const float* __restrict__ g, const float* __restrict__ beta,
                                   _Float16* __restrict__ out, long long total, int C, float invM)
{
    long long gid = (long long)blockIdx.x * blockDim.x + threadIdx.x;
    if (gid >= total) return;
    int c = (int)(gid % C);
    float m  = sums[c] * invM;
    float vv = sums[C + c] * invM - m * m;
    float s  = g[c] * rsqrtf(vv + EPSBN);
    float v  = (Y[gid] - m) * s + beta[c];
    out[gid] = (_Float16)fmaxf(v, 0.0f);
}

// BN + ReLU + segment-max scatter (atomicMax on bits; vals >= 0, dest zeroed)
__global__ void bn_relu_scatter_kernel(const float* __restrict__ Y, const float* __restrict__ sums,
                                       const float* __restrict__ g, const float* __restrict__ beta,
                                       const int* __restrict__ idx, float* __restrict__ xout,
                                       long long total, int C, float invM)
{
    long long gid = (long long)blockIdx.x * blockDim.x + threadIdx.x;
    if (gid >= total) return;
    int c = (int)(gid % C);
    long long e = gid / C;
    float m  = sums[c] * invM;
    float vv = sums[C + c] * invM - m * m;
    float s  = g[c] * rsqrtf(vv + EPSBN);
    float v  = fmaxf((Y[gid] - m) * s + beta[c], 0.0f);
    int col = idx[e];
    atomicMax((int*)&xout[(size_t)col * C + c], __float_as_int(v));
}

// Head final linear: out[N][5] = h[N][64] @ W(64x5) + b
__global__ void head_lin2_kernel(const float* __restrict__ h, const float* __restrict__ W,
                                 const float* __restrict__ b, float* __restrict__ out, int total)
{
    int gid = blockIdx.x * blockDim.x + threadIdx.x;
    if (gid >= total) return;
    int a = gid % 5, i = gid / 5;
    float s = b[a];
#pragma unroll
    for (int c = 0; c < 64; ++c) s += h[i * 64 + c] * W[c * 5 + a];
    out[gid] = s;
}

// ---------------------------------------------------------------------------
// Host orchestration
// ---------------------------------------------------------------------------
static inline unsigned int nblk(long long total) { return (unsigned int)((total + 255) / 256); }

extern "C" void kernel_launch(void* const* d_in, const int* in_sizes, int n_in,
                              void* d_out, int out_size, void* d_ws, size_t ws_size,
                              hipStream_t stream)
{
    const int N = NPTS, E = NEDGE;
    const int inch_l[3] = {64, 128, 256};
    const int ICp_l[3]  = {96, 160, 288};   // (inch+3) padded to multiple of 32
    const int OC_l[3]   = {128, 256, 512};

    auto F = [&](int i) { return (const float*)d_in[i]; };

    const float *pos, *eW, *eb, *eg, *ebt;
    const float *hW1, *hb1, *hg, *hbt, *hW2, *hb2;
    const float *l1W[3], *l1b[3], *g1[3], *b1[3], *l2W[3], *l2b[3], *g2[3], *b2[3];

    if (in_sizes[0] == N * 3) {
        // Insertion-order pytree flattening: pos first, then params.
        pos = F(0); eW = F(1); eb = F(2); eg = F(3); ebt = F(4);
        int p = 5;
        for (int i = 0; i < 3; ++i) {
            l1W[i] = F(p++); l1b[i] = F(p++); g1[i] = F(p++); b1[i] = F(p++);
            l2W[i] = F(p++); l2b[i] = F(p++); g2[i] = F(p++); b2[i] = F(p++);
        }
        hW1 = F(p++); hb1 = F(p++); hg = F(p++); hbt = F(p++); hW2 = F(p++); hb2 = F(p++);
    } else {
        // jax sorted-key flattening: params (embed, head, sa) then pos.
        ebt = F(0); eg = F(1); eW = F(2); eb = F(3);
        hbt = F(4); hg = F(5); hW1 = F(6); hb1 = F(7); hW2 = F(8); hb2 = F(9);
        int p = 10;
        for (int i = 0; i < 3; ++i) {
            b1[i] = F(p++); g1[i] = F(p++); b2[i] = F(p++); g2[i] = F(p++);
            l1W[i] = F(p++); l1b[i] = F(p++); l2W[i] = F(p++); l2b[i] = F(p++);
        }
        pos = F(p);
    }

    // Workspace layout
    char* w = (char*)d_ws;
    int*      idx  = (int*)w;       w += (size_t)E * 4;
    float*    sums = (float*)w;     w += 4096;                       // up to 2*512 f32
    _Float16* Wt   = (_Float16*)w;  w += (size_t)512 * 288 * 2;      // max lin1 weight tile (transposed)
    float*    x0   = (float*)w;     w += (size_t)N * 512 * 4;
    float*    x1   = (float*)w;     w += (size_t)N * 512 * 4;
    _Float16* Ah   = (_Float16*)w;  w += (size_t)E * 288 * 2;        // packed edge feats
    _Float16* Hh   = (_Float16*)w;  w += (size_t)E * 512 * 2;        // hidden f16
    float*    Y    = (float*)w;     w += (size_t)E * 512 * 4;        // GEMM out (reused)

    // 1) kNN graph
    knn_kernel<<<N / 256, 256, 0, stream>>>(pos, idx);

    // 2) Embed: linear(3->64) + BN + ReLU -> x0 (f32)
    {
        long long tot = (long long)N * 64;
        embed_kernel<<<nblk(tot), 256, 0, stream>>>(pos, eW, eb, Y, (int)tot);
        zero_kernel<<<1, 256, 0, stream>>>(sums, 128);
        bn_stats_kernel<<<nblk((long long)64 * ((N + 255) / 256)), 256, 0, stream>>>(Y, sums, N, 64);
        bn_relu_f32_kernel<<<nblk(tot), 256, 0, stream>>>(Y, sums, eg, ebt, x0, tot, 64, 1.0f / N);
    }

    // 3) SA layers
    float* xin = x0;
    float* xout = x1;
    for (int i = 0; i < 3; ++i) {
        const int inch = inch_l[i], ICp = ICp_l[i], OC = OC_l[i];
        const float invE = 1.0f / (float)E;

        // lin1: pack edge feats, convert+transpose W, GEMM
        cvt_wT_f16_kernel<<<nblk((long long)OC * ICp), 256, 0, stream>>>(
            l1W[i], Wt, inch + 3, ICp, OC, (long long)OC * ICp);
        pack_sa_kernel<<<nblk((long long)E * ICp), 256, 0, stream>>>(
            xin, pos, idx, Ah, inch, ICp, (long long)E * ICp);
        gemm_wmma_kernel<<<dim3(E / 128, OC / 64), 256, 0, stream>>>(Ah, Wt, l1b[i], Y, E, ICp, OC);

        // bn1 + relu -> f16
        zero_kernel<<<1, 256, 0, stream>>>(sums, 2 * OC);
        bn_stats_kernel<<<nblk((long long)OC * ((E + 255) / 256)), 256, 0, stream>>>(Y, sums, E, OC);
        bn_relu_f16_kernel<<<nblk((long long)E * OC), 256, 0, stream>>>(
            Y, sums, g1[i], b1[i], Hh, (long long)E * OC, OC, invE);

        // lin2: GEMM (OC x OC)
        cvt_wT_f16_kernel<<<nblk((long long)OC * OC), 256, 0, stream>>>(
            l2W[i], Wt, OC, OC, OC, (long long)OC * OC);
        gemm_wmma_kernel<<<dim3(E / 128, OC / 64), 256, 0, stream>>>(Hh, Wt, l2b[i], Y, E, OC, OC);

        // bn2 + relu + segment-max scatter into xout
        zero_kernel<<<1, 256, 0, stream>>>(sums, 2 * OC);
        bn_stats_kernel<<<nblk((long long)OC * ((E + 255) / 256)), 256, 0, stream>>>(Y, sums, E, OC);
        zero_kernel<<<nblk((long long)N * OC), 256, 0, stream>>>(xout, (long long)N * OC);
        bn_relu_scatter_kernel<<<nblk((long long)E * OC), 256, 0, stream>>>(
            Y, sums, g2[i], b2[i], idx, xout, (long long)E * OC, OC, invE);

        float* t = xin; xin = xout; xout = t;
    }

    // 4) Head: lin1 (512->64) via WMMA, BN+ReLU, lin2 (64->5) scalar
    {
        cvt_f16_kernel<<<nblk((long long)N * 512), 256, 0, stream>>>(
            xin, Ah, (long long)N * 512);                             // activations f32->f16
        cvt_wT_f16_kernel<<<nblk((long long)64 * 512), 256, 0, stream>>>(
            hW1, Wt, 512, 512, 64, (long long)64 * 512);
        gemm_wmma_kernel<<<dim3(N / 128, 1), 256, 0, stream>>>(Ah, Wt, hb1, Y, N, 512, 64);

        zero_kernel<<<1, 256, 0, stream>>>(sums, 128);
        bn_stats_kernel<<<nblk((long long)64 * ((N + 255) / 256)), 256, 0, stream>>>(Y, sums, N, 64);
        bn_relu_f32_kernel<<<nblk((long long)N * 64), 256, 0, stream>>>(
            Y, sums, hg, hbt, xout, (long long)N * 64, 64, 1.0f / N);

        head_lin2_kernel<<<nblk((long long)N * 5), 256, 0, stream>>>(
            xout, hW2, hb2, (float*)d_out, N * 5);
    }
}
